// Attention_42528766165493
// MI455X (gfx1250) — matrix operand order repaired and checked by reading, not compiled, not run on hardware
//
#include <hip/hip_runtime.h>
#include <hip/hip_bf16.h>
#include <stdint.h>

// ---------------------------------------------------------------------------
// MI455X (gfx1250) fused causal attention, bf16 WMMA pipeline, f32 accumulate.
// ---------------------------------------------------------------------------

typedef __attribute__((ext_vector_type(16))) __bf16 v16bf;
typedef __attribute__((ext_vector_type(8)))  float  v8f;

union FragU {
    uint4 u4[2];
    v16bf v;
};

__device__ __forceinline__ __bf16 to_bf16(float f) {
    unsigned int u = __builtin_bit_cast(unsigned int, f);
    u += 0x7FFFu + ((u >> 16) & 1u);          // round-to-nearest-even
    unsigned short h = (unsigned short)(u >> 16);
    return __builtin_bit_cast(__bf16, h);
}

__device__ __forceinline__ v8f wmma_bf16(v16bf a, v16bf b, v8f c) {
    // (neg_a, A, neg_b, B, c_mod, C, reuse_a, reuse_b)
    return __builtin_amdgcn_wmma_f32_16x16x32_bf16(false, a, false, b,
                                                   (short)0, c, false, false);
}

// A-fragment (16x32 bf16). ISA layout: lanes 0-15 hold K{0-7,16-23},
// lanes 16-31 hold K{8-15,24-31}; row = lane%16.  Two contiguous 16B loads.
__device__ __forceinline__ v16bf load_frag_a(const __bf16* base, int row,
                                             int rowStrideBytes, int half,
                                             int kByteOff) {
    FragU f;
    const char* p = (const char*)base + row * rowStrideBytes + kByteOff + half * 16;
    f.u4[0] = *(const uint4*)(p);
    f.u4[1] = *(const uint4*)(p + 32);
    return f.v;
}

// B-fragment (32x16 bf16) from Bt storage (rows = output columns, k contiguous).
// ISA layout: lanes 0-15 hold K{0-7,16-23}, lanes 16-31 hold K{8-15,24-31}; col = lane%16.
__device__ __forceinline__ v16bf load_frag_b(const __bf16* base, int row,
                                             int rowStrideBytes, int half,
                                             int kByteOff) {
    FragU f;
    const char* p = (const char*)base + row * rowStrideBytes + kByteOff + half * 16;
    f.u4[0] = *(const uint4*)(p);
    f.u4[1] = *(const uint4*)(p + 32);
    return f.v;
}

// ---------------------------------------------------------------------------
// fp32 -> bf16 elementwise convert
// ---------------------------------------------------------------------------
__global__ __launch_bounds__(256) void cvt_bf16_kernel(const float* __restrict__ in,
                                                       __bf16* __restrict__ out, int n) {
    for (int i = blockIdx.x * blockDim.x + threadIdx.x; i < n;
         i += gridDim.x * blockDim.x)
        out[i] = to_bf16(in[i]);
}

// ---------------------------------------------------------------------------
// W[K,N] fp32 -> WT[N,K] bf16 (LDS tiled transpose, 32x32 tiles)
// ---------------------------------------------------------------------------
__global__ __launch_bounds__(256) void transpose_bf16_kernel(const float* __restrict__ W,
                                                             __bf16* __restrict__ WT,
                                                             int K, int N) {
    __shared__ float tile[32][33];
    const int bx = blockIdx.x * 32;  // N
    const int by = blockIdx.y * 32;  // K
    const int tx = threadIdx.x, ty = threadIdx.y;  // 32 x 8
    for (int i = ty; i < 32; i += 8)
        tile[i][tx] = W[(size_t)(by + i) * N + (bx + tx)];
    __syncthreads();
    for (int i = ty; i < 32; i += 8)
        WT[(size_t)(bx + i) * K + (by + tx)] = to_bf16(tile[tx][i]);
}

// ---------------------------------------------------------------------------
// bf16 GEMM: C[M,N] = A[M,1024] * Bt[N,1024]^T + bias
// Block tile 128x64, 256 threads (8 waves), 32x32 per wave, K-step 32 via LDS.
// MODE 0: QKV epilogue (Q,K row-major bf16; V stored transposed per column).
// MODE 1: fp32 output epilogue.
// ---------------------------------------------------------------------------
template <int MODE>
__global__ __launch_bounds__(256) void gemm_bf16_kernel(
    const __bf16* __restrict__ A, const __bf16* __restrict__ Bt,
    const float* __restrict__ bias,
    __bf16* __restrict__ Qb, __bf16* __restrict__ Kb, __bf16* __restrict__ Vt,
    float* __restrict__ outF) {
    constexpr int KD = 1024;
    const int tid  = threadIdx.x;
    const int w    = tid >> 5;
    const int lane = tid & 31, l16 = lane & 15, half = lane >> 4;
    const int wm = w >> 1, wn = w & 1;          // 4 x 2 wave grid
    const int m0 = blockIdx.x * 128, n0 = blockIdx.y * 64;

    __shared__ alignas(16) __bf16 As[128 * 32];  // 8 KB
    __shared__ alignas(16) __bf16 Bs[64 * 32];   // 4 KB

    v8f acc[2][2];
    for (int i = 0; i < 2; i++)
        for (int j = 0; j < 2; j++)
            for (int e = 0; e < 8; e++) acc[i][j][e] = 0.f;

    const uint4* Ag = (const uint4*)A;
    const uint4* Bg = (const uint4*)Bt;

    for (int k0 = 0; k0 < KD; k0 += 32) {
        // cooperative A tile: 128 rows x 64B
        for (int idx = tid; idx < 512; idx += 256) {
            int r = idx >> 2, c = idx & 3;
            ((uint4*)As)[idx] = Ag[((size_t)(m0 + r) * KD + k0) / 8 + c];
        }
        // cooperative B tile: 64 rows x 64B (one uint4 per thread)
        {
            int r = tid >> 2, c = tid & 3;
            ((uint4*)Bs)[tid] = Bg[((size_t)(n0 + r) * KD + k0) / 8 + c];
        }
        __syncthreads();

        v16bf af[2], bfq[2];
        for (int mt = 0; mt < 2; mt++)
            af[mt] = load_frag_a(As, wm * 32 + mt * 16 + l16, 64, half, 0);
        for (int nt = 0; nt < 2; nt++)
            bfq[nt] = load_frag_b(Bs, wn * 32 + nt * 16 + l16, 64, half, 0);
        for (int mt = 0; mt < 2; mt++)
            for (int nt = 0; nt < 2; nt++)
                acc[mt][nt] = wmma_bf16(af[mt], bfq[nt], acc[mt][nt]);
        __syncthreads();
    }

    // C layout: lane -> col (lane%16), VGPR j -> row j + 8*(lane/16)
    for (int mt = 0; mt < 2; mt++)
        for (int nt = 0; nt < 2; nt++)
            for (int j = 0; j < 8; j++) {
                int row = m0 + wm * 32 + mt * 16 + j + 8 * half;
                int col = n0 + wn * 32 + nt * 16 + l16;
                float v = acc[mt][nt][j] + bias[col];
                if (MODE == 0) {
                    if (col < 1024)
                        Qb[(size_t)row * 1024 + col] = to_bf16(v);
                    else if (col < 2048)
                        Kb[(size_t)row * 1024 + (col - 1024)] = to_bf16(v);
                    else  // V stored transposed: Vt[vcol][s]
                        Vt[(size_t)(col - 2048) * 4096 + row] = to_bf16(v);
                } else {
                    outF[(size_t)row * 1024 + col] = v;
                }
            }
}

// ---------------------------------------------------------------------------
// Fused causal flash attention, one head x 64 query rows per block.
// 128 threads = 4 waves; wave w owns query rows [qb*64 + w*16, +16).
// ---------------------------------------------------------------------------
__global__ __launch_bounds__(128) void attn_kernel(
    const __bf16* __restrict__ Q, const __bf16* __restrict__ Kg,
    const __bf16* __restrict__ Vt, __bf16* __restrict__ Ab) {
    constexpr int S = 4096, D = 1024, DH = 64;
    const int qb = blockIdx.x, h = blockIdx.y;
    const int tid  = threadIdx.x;
    const int w    = tid >> 5;
    const int lane = tid & 31, l16 = lane & 15, half = lane >> 4;

    __shared__ alignas(16) __bf16 Kt[64 * 64];      // 8 KB  (keys x dh)
    __shared__ alignas(16) __bf16 Vs[64 * 64];      // 8 KB  (dh x keys)
    __shared__ alignas(16) __bf16 Ps[4][16 * 64];   // 8 KB  (per-wave P tile)

    // Q fragments (16 rows x 64 dh) held in registers for the whole kernel
    v16bf qf[2];
    {
        const int qrow = qb * 64 + w * 16 + l16;
        const char* p = (const char*)(Q + (size_t)qrow * D + h * DH);
        for (int kc = 0; kc < 2; kc++) {
            FragU f;
            f.u4[0] = *(const uint4*)(p + kc * 64 + half * 16);
            f.u4[1] = *(const uint4*)(p + kc * 64 + half * 16 + 32);
            qf[kc] = f.v;
        }
    }

    float m_run[8], l_run[8];
    v8f o[4];
    for (int j = 0; j < 8; j++) { m_run[j] = -1e30f; l_run[j] = 0.f; }
    for (int t = 0; t < 4; t++)
        for (int e = 0; e < 8; e++) o[t][e] = 0.f;

    const float scale = 0.125f;  // 1/sqrt(64)

    for (int kb = 0; kb <= qb; kb++) {
        // Stage K tile (64 keys x 64 dh) and V^T tile (64 dh x 64 keys) in LDS
        for (int idx = tid; idx < 512; idx += 128) {
            int r = idx >> 3, c = idx & 7;
            ((uint4*)Kt)[idx] = *(const uint4*)((const char*)Kg +
                ((size_t)(kb * 64 + r) * D + h * DH) * 2 + c * 16);
            ((uint4*)Vs)[idx] = *(const uint4*)((const char*)Vt +
                ((size_t)(h * DH + r) * S + kb * 64) * 2 + c * 16);
        }
        if (kb < qb) {  // prefetch next K tile while this block computes
            const char* nk = (const char*)Kg +
                ((size_t)((kb + 1) * 64 + (tid >> 1)) * D + h * DH) * 2 + (tid & 1) * 64;
            __builtin_prefetch(nk, 0, 1);
        }
        __syncthreads();

        // scores S = Q * K^T : 4 key tiles x 2 k-chunks (dh) = 8 WMMAs
        v8f s[4];
        for (int t = 0; t < 4; t++) {
            v8f a;
            for (int e = 0; e < 8; e++) a[e] = 0.f;
            for (int kc = 0; kc < 2; kc++) {
                v16bf kf = load_frag_b(Kt, t * 16 + l16, 128, half, kc * 64);
                a = wmma_bf16(qf[kc], kf, a);
            }
            s[t] = a;
        }

        // scale + causal mask + running row-max (C layout: col = lane%16)
        const bool diag = (kb == qb);
        float mnew[8];
        for (int j = 0; j < 8; j++) mnew[j] = -1e30f;
        for (int t = 0; t < 4; t++) {
            const int key = kb * 64 + t * 16 + l16;
            for (int j = 0; j < 8; j++) {
                float v = s[t][j] * scale;
                const int qrow = qb * 64 + w * 16 + j + 8 * half;
                if (diag && key > qrow) v = -1e30f;
                s[t][j] = v;
                mnew[j] = fmaxf(mnew[j], v);
            }
        }
        for (int j = 0; j < 8; j++)
            for (int m = 1; m < 16; m <<= 1)
                mnew[j] = fmaxf(mnew[j], __shfl_xor(mnew[j], m, 32));

        float alpha[8];
        for (int j = 0; j < 8; j++) {
            float m2 = fmaxf(m_run[j], mnew[j]);
            alpha[j] = __expf(m_run[j] - m2);
            m_run[j] = m2;
        }

        // P = exp(S - m), row sums, stash bf16 P in per-wave LDS
        float rs[8];
        for (int j = 0; j < 8; j++) rs[j] = 0.f;
        for (int t = 0; t < 4; t++)
            for (int j = 0; j < 8; j++) {
                float p = __expf(s[t][j] - m_run[j]);
                rs[j] += p;
                Ps[w][(j + 8 * half) * 64 + t * 16 + l16] = to_bf16(p);
            }
        for (int j = 0; j < 8; j++)
            for (int m = 1; m < 16; m <<= 1)
                rs[j] += __shfl_xor(rs[j], m, 32);
        for (int j = 0; j < 8; j++) l_run[j] = l_run[j] * alpha[j] + rs[j];
        for (int t = 0; t < 4; t++)
            for (int j = 0; j < 8; j++) o[t][j] *= alpha[j];

        // O += P * V : P as A-fragments (same-wave LDS round-trip), 8 WMMAs
        v16bf pf[2];
        for (int kc = 0; kc < 2; kc++)
            pf[kc] = load_frag_a(&Ps[w][0], l16, 128, half, kc * 64);
        for (int t = 0; t < 4; t++)
            for (int kc = 0; kc < 2; kc++) {
                v16bf vf = load_frag_b(Vs, t * 16 + l16, 128, half, kc * 64);
                o[t] = wmma_bf16(pf[kc], vf, o[t]);
            }
        __syncthreads();
    }

    // epilogue: O / l -> bf16 merged-head layout
    for (int t = 0; t < 4; t++)
        for (int j = 0; j < 8; j++) {
            const int qrow = qb * 64 + w * 16 + j + 8 * half;
            const int col  = h * DH + t * 16 + l16;
            Ab[(size_t)qrow * D + col] = to_bf16(o[t][j] / l_run[j]);
        }
}

// ---------------------------------------------------------------------------
// Launch
// ---------------------------------------------------------------------------
extern "C" void kernel_launch(void* const* d_in, const int* in_sizes, int n_in,
                              void* d_out, int out_size, void* d_ws, size_t ws_size,
                              hipStream_t stream) {
    constexpr int S = 4096, D = 1024, H = 16;
    const float* x      = (const float*)d_in[0];
    // d_in[1] = additive causal mask (re-derived analytically in-kernel)
    const float* W_attn = (const float*)d_in[2];
    const float* b_attn = (const float*)d_in[3];
    const float* W_proj = (const float*)d_in[4];
    const float* b_proj = (const float*)d_in[5];
    float* out = (float*)d_out;

    char* ws = (char*)d_ws;
    __bf16* xb  = (__bf16*)ws;                 ws += (size_t)S * D * 2;       // 8 MiB
    __bf16* WaT = (__bf16*)ws;                 ws += (size_t)3 * D * D * 2;   // 6 MiB
    __bf16* WpT = (__bf16*)ws;                 ws += (size_t)D * D * 2;       // 2 MiB
    __bf16* Qb  = (__bf16*)ws;                 ws += (size_t)S * D * 2;
    __bf16* Kb  = (__bf16*)ws;                 ws += (size_t)S * D * 2;
    __bf16* Vt  = (__bf16*)ws;                 ws += (size_t)S * D * 2;
    __bf16* Ab  = (__bf16*)ws;                 ws += (size_t)S * D * 2;

    // 1) fp32 -> bf16 activations, transposed bf16 weights
    cvt_bf16_kernel<<<2048, 256, 0, stream>>>(x, xb, S * D);
    transpose_bf16_kernel<<<dim3(3 * D / 32, D / 32), dim3(32, 8), 0, stream>>>(
        W_attn, WaT, D, 3 * D);
    transpose_bf16_kernel<<<dim3(D / 32, D / 32), dim3(32, 8), 0, stream>>>(
        W_proj, WpT, D, D);

    // 2) QKV = x @ W_attn + b  (V stored transposed for the P@V B-operand)
    gemm_bf16_kernel<0><<<dim3(S / 128, 3 * D / 64), 256, 0, stream>>>(
        xb, WaT, b_attn, Qb, Kb, Vt, nullptr);

    // 3) fused causal flash attention
    attn_kernel<<<dim3(S / 64, H), 128, 0, stream>>>(Qb, Kb, Vt, Ab);

    // 4) out = A @ W_proj + b  (fp32 epilogue)
    gemm_bf16_kernel<1><<<dim3(S / 128, D / 64), 256, 0, stream>>>(
        Ab, WpT, b_proj, nullptr, nullptr, nullptr, out);
}
